// DGCNNpp_54829552501121
// MI455X (gfx1250) — matrix operand order repaired and checked by reading, not compiled, not run on hardware
//
#include <hip/hip_runtime.h>
#include <cstdint>

typedef __attribute__((ext_vector_type(16))) _Float16 v16h;
typedef __attribute__((ext_vector_type(8)))  float    v8f;

__device__ __forceinline__ float leaky_f(float x){ return x >= 0.f ? x : 0.2f*x; }

// ---------------- prep: x [8,3,1024] -> xt32 [8][1024][3], xt16 [8][1024][32] (zero padded) ------
__global__ void k_prep_x(const float* __restrict__ x, float* __restrict__ xt32,
                         _Float16* __restrict__ xt16){
  int t = blockIdx.x*blockDim.x + threadIdx.x;
  if (t >= 8*1024) return;
  int b = t >> 10, n = t & 1023;
  const float* src = x + (size_t)b*3*1024 + n;
  float v0 = src[0], v1 = src[1024], v2 = src[2048];
  float* d32 = xt32 + (size_t)t*3;
  d32[0]=v0; d32[1]=v1; d32[2]=v2;
  _Float16* d16 = xt16 + (size_t)t*32;
  #pragma unroll
  for (int c=0;c<32;c++) d16[c] = (_Float16)0.f;
  d16[0]=(_Float16)v0; d16[1]=(_Float16)v1; d16[2]=(_Float16)v2;
}

// ------- pack combined weights, f16, rows: [WtA | Wg1A | Wa1A | WtB-WtA | Wg1B-Wg1A | Wa1B-Wa1A]
__global__ void k_prep_w(const float* __restrict__ Wt, const float* __restrict__ Wg1,
                         const float* __restrict__ Wa1, _Float16* __restrict__ A,
                         int Cout, int C, int Kp){
  int o2 = Cout + 16;
  int total = 2*o2*Kp;
  int t = blockIdx.x*blockDim.x + threadIdx.x;
  if (t >= total) return;
  int r = t / Kp, c = t % Kp;
  float v = 0.f;
  if (c < C){
    int hb = (r >= o2) ? 1 : 0;
    int rr = r - hb*o2;
    const float* W; int row;
    if (rr < Cout)        { W = Wt;  row = rr; }
    else if (rr < Cout+8) { W = Wg1; row = rr - Cout; }
    else                  { W = Wa1; row = rr - Cout - 8; }
    float a = W[row*(2*C) + c];
    v = hb ? (W[row*(2*C) + C + c] - a) : a;
  }
  A[t] = (_Float16)v;
}

__global__ void k_prep_w5(const float* __restrict__ W5, _Float16* __restrict__ A){
  int t = blockIdx.x*blockDim.x + threadIdx.x;
  if (t < 1024*512) A[t] = (_Float16)W5[t];
}

// ---------------- kNN: one wave per point, LDS distance row, 20 max-extractions ----------------
__global__ void k_knn(const float* __restrict__ X, int C, int* __restrict__ idx){
  __shared__ float pd[8][1024];
  __shared__ float xn[8][128];
  int wave = threadIdx.x >> 5, lane = threadIdx.x & 31;
  int p = blockIdx.x*8 + wave;
  int b = p >> 10, n = p & 1023;
  const float* Xb = X + (size_t)b*1024*C;
  for (int c = lane; c < C; c += 32) xn[wave][c] = Xb[(size_t)n*C + c];
  __syncthreads();
  for (int m = lane; m < 1024; m += 32){
    const float* xm = Xb + (size_t)m*C;
    float acc = 0.f;
    for (int c=0;c<C;c++){ float d = xn[wave][c] - xm[c]; acc -= d*d; }
    pd[wave][m] = acc;
  }
  __syncthreads();
  for (int t=0;t<20;t++){
    float bv = -3.0e38f; int bi = 0x7fffffff;
    for (int m = lane; m < 1024; m += 32){
      float v = pd[wave][m];
      if (v > bv){ bv = v; bi = m; }      // strict > keeps lowest index within lane
    }
    #pragma unroll
    for (int off=16; off>=1; off>>=1){
      float ov = __shfl_xor(bv, off, 32);
      int   oi = __shfl_xor(bi, off, 32);
      if (ov > bv || (ov == bv && oi < bi)){ bv = ov; bi = oi; }
    }
    if (lane == 0) idx[(size_t)p*20 + t] = bi;
    if (lane == (bi & 31)) pd[wave][bi] = -3.0e38f;
    __syncthreads();
  }
}

// ------------- WMMA GEMM, 2x4 register blocking: wave computes a 32(M) x 64(N) tile -----------
// Dt[b][n][m] = sum_k A[m][k] * Bt[b][n][k]; A row-major [M x Kp], Bt row stride ldb, N = 1024.
__global__ void k_gemm(const _Float16* __restrict__ A, const _Float16* __restrict__ Bt,
                       float* __restrict__ Dt, int M, int Kp, int ldb){
  const int N = 1024;
  int bb = blockIdx.y;
  int tilesM = M >> 5;                 // 32-row macro tiles
  int tiles  = tilesM * (N >> 6);      // 64-col macro tiles
  int w = blockIdx.x*8 + (threadIdx.x >> 5);
  if (w >= tiles) return;
  int mt = w % tilesM, nt = w / tilesM;
  int m0 = mt << 5, n0 = nt << 6;
  int lane = threadIdx.x & 31;
  int hf = lane >> 4, r = lane & 15;
  const _Float16* Arow0 = A  + (size_t)(m0 + r)*Kp;
  const _Float16* Arow1 = Arow0 + (size_t)16*Kp;
  const _Float16* Brow0 = Bt + ((size_t)bb*N + n0 + r)*ldb;
  const _Float16* Brow1 = Brow0 + (size_t)16*ldb;
  const _Float16* Brow2 = Brow0 + (size_t)32*ldb;
  const _Float16* Brow3 = Brow0 + (size_t)48*ldb;
  v8f acc[2][4];
  #pragma unroll
  for (int i=0;i<2;i++)
    #pragma unroll
    for (int j=0;j<4;j++) acc[i][j] = (v8f){};
  union F { v16h h; uint32_t u[8]; };
  for (int k0 = 0; k0 < Kp; k0 += 32){
    F a0, a1, b0, b1, b2, b3;
    #pragma unroll
    for (int v=0; v<8; v++){
      int kkA = (v<4 ? 2*v : 2*v+8) + 8*hf;   // A 16x32 f16 layout (ISA 7.12.2)
      a0.u[v] = *(const uint32_t*)(Arow0 + k0 + kkA);
      a1.u[v] = *(const uint32_t*)(Arow1 + k0 + kkA);
      int kkB = (v<4 ? 2*v : 2*v+8) + 8*hf;   // B 32x16 f16 layout
      b0.u[v] = *(const uint32_t*)(Brow0 + k0 + kkB);
      b1.u[v] = *(const uint32_t*)(Brow1 + k0 + kkB);
      b2.u[v] = *(const uint32_t*)(Brow2 + k0 + kkB);
      b3.u[v] = *(const uint32_t*)(Brow3 + k0 + kkB);
    }
    acc[0][0] = __builtin_amdgcn_wmma_f32_16x16x32_f16(false, a0.h, false, b0.h, (short)0, acc[0][0], false, false);
    acc[1][0] = __builtin_amdgcn_wmma_f32_16x16x32_f16(false, a1.h, false, b0.h, (short)0, acc[1][0], false, false);
    acc[0][1] = __builtin_amdgcn_wmma_f32_16x16x32_f16(false, a0.h, false, b1.h, (short)0, acc[0][1], false, false);
    acc[1][1] = __builtin_amdgcn_wmma_f32_16x16x32_f16(false, a1.h, false, b1.h, (short)0, acc[1][1], false, false);
    acc[0][2] = __builtin_amdgcn_wmma_f32_16x16x32_f16(false, a0.h, false, b2.h, (short)0, acc[0][2], false, false);
    acc[1][2] = __builtin_amdgcn_wmma_f32_16x16x32_f16(false, a1.h, false, b2.h, (short)0, acc[1][2], false, false);
    acc[0][3] = __builtin_amdgcn_wmma_f32_16x16x32_f16(false, a0.h, false, b3.h, (short)0, acc[0][3], false, false);
    acc[1][3] = __builtin_amdgcn_wmma_f32_16x16x32_f16(false, a1.h, false, b3.h, (short)0, acc[1][3], false, false);
  }
  // D element (v, lane) of tile (i,j) = row m0+16i+v+8*hf, col n0+16j+r -> transposed store,
  // 8 contiguous floats per lane.
  #pragma unroll
  for (int j=0;j<4;j++){
    size_t row = ((size_t)bb*N + n0 + 16*j + r)*M;
    *(v8f*)(Dt + row + m0      + 8*hf) = acc[0][j];
    *(v8f*)(Dt + row + m0 + 16 + 8*hf) = acc[1][j];
  }
}

// ---------------- per-point combine: BN/leaky, gate, attn softmax, sum+max aggregation ---------
__global__ void k_combine(const float* __restrict__ Zt, const int* __restrict__ idx,
                          const float* __restrict__ gt,  const float* __restrict__ bt,
                          const float* __restrict__ gg,  const float* __restrict__ bgn,
                          const float* __restrict__ Wg2, const float* __restrict__ bg2,
                          const float* __restrict__ ga,  const float* __restrict__ ban,
                          const float* __restrict__ Wa2, const float* __restrict__ ba2,
                          int Cout, int Mtot, float* __restrict__ h32,
                          _Float16* __restrict__ cat, int cat_off){
  __shared__ float gate_s[20], w_s[20], logit_s[20];
  __shared__ int   idx_s[20];
  int p = blockIdx.x;
  int b = p >> 10;
  int o2 = Mtot >> 1;
  const float* zn = Zt + (size_t)p*Mtot;
  const float inv = rsqrtf(1.f + 1e-5f);
  int tid = threadIdx.x;
  if (tid < 20){
    int j = idx[(size_t)p*20 + tid];
    idx_s[tid] = j;
    const float* zj = Zt + (size_t)(b*1024 + j)*Mtot;
    float gd = 0.f, ad = 0.f;
    #pragma unroll
    for (int q=0;q<8;q++){
      float gv = zj[Cout+q]   + zn[o2+Cout+q];
      gv = leaky_f(gv * (gg[q]*inv) + bgn[q]);
      gd += Wg2[q]*gv;
      float av = zj[Cout+8+q] + zn[o2+Cout+8+q];
      av = leaky_f(av * (ga[q]*inv) + ban[q]);
      ad += Wa2[q]*av;
    }
    gate_s[tid]  = 1.f/(1.f + expf(-(gd + bg2[0])));
    logit_s[tid] = ad + ba2[0];
  }
  __syncthreads();
  if (tid == 0){
    float mx = -3e38f;
    for (int j=0;j<20;j++) mx = fmaxf(mx, logit_s[j]);
    float s = 0.f;
    for (int j=0;j<20;j++){ float e = expf(logit_s[j]-mx); w_s[j] = e; s += e; }
    float rs = 1.f/s;
    for (int j=0;j<20;j++) w_s[j] *= rs;
  }
  __syncthreads();
  for (int i = tid; i < Cout; i += blockDim.x){
    float zb = zn[o2 + i];
    float st = gt[i]*inv, bbias = bt[i];
    float s = 0.f, mx = -3e38f;
    #pragma unroll 4
    for (int j=0;j<20;j++){
      float tv = Zt[(size_t)(b*1024 + idx_s[j])*Mtot + i] + zb;
      tv = leaky_f(tv*st + bbias);
      float cl = tv * gate_s[j];
      s += cl * w_s[j];
      mx = fmaxf(mx, cl);
    }
    float o = s + mx;
    h32[(size_t)p*Cout + i] = o;
    cat[(size_t)p*512 + cat_off + i] = (_Float16)o;
  }
}

// ---------------- final BN + leaky + max over N ----------------
__global__ void k_reduce(const float* __restrict__ Zt5, const float* __restrict__ g5,
                         const float* __restrict__ b5, float* __restrict__ out){
  int b = blockIdx.x >> 2;
  int m = ((blockIdx.x & 3) << 8) + threadIdx.x;
  const float inv = rsqrtf(1.f + 1e-5f);
  float sc = g5[m]*inv, bi = b5[m];
  float mx = -3e38f;
  const float* base = Zt5 + (size_t)b*1024*1024 + m;
  for (int n=0;n<1024;n++){
    float v = base[(size_t)n*1024];
    v = leaky_f(v*sc + bi);
    mx = fmaxf(mx, v);
  }
  out[b*1024 + m] = mx;
}

extern "C" void kernel_launch(void* const* d_in, const int* in_sizes, int n_in,
                              void* d_out, int out_size, void* d_ws, size_t ws_size,
                              hipStream_t stream){
  (void)in_sizes; (void)n_in; (void)out_size; (void)ws_size;
  const float* x = (const float*)d_in[0];
  auto P = [&](int s, int k){ return (const float*)d_in[1 + s*13 + k]; };
  const float* W5 = (const float*)d_in[53];
  const float* g5 = (const float*)d_in[54];
  const float* b5 = (const float*)d_in[55];

  char* ws = (char*)d_ws;
  size_t off = 0;
  auto alloc = [&](size_t bytes){ void* p = ws + off; off = (off + bytes + 255) & ~(size_t)255; return p; };
  float*    xt32 = (float*)   alloc((size_t)8*1024*3*4);
  _Float16* xt16 = (_Float16*)alloc((size_t)8*1024*32*2);
  _Float16* cat  = (_Float16*)alloc((size_t)8*1024*512*2);
  float*    h32  = (float*)   alloc((size_t)8*1024*256*4);
  int*      idx  = (int*)     alloc((size_t)8*1024*20*4);
  float*    Zt   = (float*)   alloc((size_t)8*1024*1024*4);
  _Float16* Aw   = (_Float16*)alloc((size_t)1024*512*2);

  hipLaunchKernelGGL(k_prep_x, dim3(32), dim3(256), 0, stream, x, xt32, xt16);

  const int Couts[4]  = {64, 64, 128, 256};
  const int Cs[4]     = {3, 64, 64, 128};
  const int Kps[4]    = {32, 64, 64, 128};
  const int catoff[4] = {0, 64, 128, 256};

  for (int s=0;s<4;s++){
    int Cout = Couts[s], C = Cs[s], Kp = Kps[s];
    int Mtot = 2*(Cout + 16);
    const float* kin = (s==0) ? xt32 : h32;
    hipLaunchKernelGGL(k_knn, dim3(1024), dim3(256), 0, stream, kin, C, idx);
    int tot = Mtot*Kp;
    hipLaunchKernelGGL(k_prep_w, dim3((tot+255)/256), dim3(256), 0, stream,
                       P(s,0), P(s,3), P(s,8), Aw, Cout, C, Kp);
    const _Float16* Bt = (s==0) ? xt16 : (cat + catoff[s-1]);
    int ldb = (s==0) ? 32 : 512;
    int tiles = (Mtot/32)*16;      // (M/32) * (N/64)
    hipLaunchKernelGGL(k_gemm, dim3((tiles+7)/8, 8), dim3(256), 0, stream,
                       Aw, Bt, Zt, Mtot, Kp, ldb);
    hipLaunchKernelGGL(k_combine, dim3(8192), dim3(128), 0, stream, Zt, idx,
                       P(s,1), P(s,2), P(s,4), P(s,5), P(s,6), P(s,7),
                       P(s,9), P(s,10), P(s,11), P(s,12),
                       Cout, Mtot, h32, cat, catoff[s]);
  }

  hipLaunchKernelGGL(k_prep_w5, dim3(2048), dim3(256), 0, stream, W5, Aw);
  int tiles = 32*16;               // (1024/32) * (1024/64)
  hipLaunchKernelGGL(k_gemm, dim3((tiles+7)/8, 8), dim3(256), 0, stream,
                     Aw, cat, Zt, 1024, 512, 512);
  hipLaunchKernelGGL(k_reduce, dim3(32), dim3(256), 0, stream, Zt, g5, b5, (float*)d_out);
}
